// ExpertGNN_17128329576802
// MI455X (gfx1250) — matrix operand order repaired and checked by reading, not compiled, not run on hardware
//
#include <hip/hip_runtime.h>

#define N_NODES   20000
#define N_EDGES   320000
#define F_IN      128
#define HGAT      8
#define F_GAT     128
#define HF        (HGAT * F_GAT)     // 1024
#define NCLS      10
#define NEG_SLOPE 0.2f
#define EPS_GAT   1e-16f
#define E_TOT     (N_EDGES + N_NODES) // logical edges incl. self loops

typedef __attribute__((ext_vector_type(16))) __bf16 v16bf;
typedef __attribute__((ext_vector_type(8)))  __bf16 v8bf;
typedef __attribute__((ext_vector_type(8)))  float  v8f;

// ---------- helpers ----------
__device__ __forceinline__ unsigned short f2bf(float f) {
  unsigned u = __float_as_uint(f);
  u += 0x7FFFu + ((u >> 16) & 1u);          // round-to-nearest-even
  return (unsigned short)(u >> 16);
}
__device__ __forceinline__ unsigned f2ord(float f) {   // monotone float->uint
  unsigned b = __float_as_uint(f);
  return (b & 0x80000000u) ? ~b : (b | 0x80000000u);
}
__device__ __forceinline__ float ord2f(unsigned u) {
  unsigned b = (u & 0x80000000u) ? (u & 0x7FFFFFFFu) : ~u;
  return __uint_as_float(b);
}
__device__ __forceinline__ void edge_sd(const long long* __restrict__ ei,
                                        int e, int& s, int& d) {
  if (e < N_EDGES) { s = (int)ei[e]; d = (int)ei[N_EDGES + e]; }
  else             { s = e - N_EDGES; d = s; }          // self loop
}

// ---------- generic fills / conversions ----------
__global__ void fill_f32(float* __restrict__ p, int n, float v) {
  int t = blockIdx.x * blockDim.x + threadIdx.x;
  if (t < n) p[t] = v;
}
__global__ void fill_u32(unsigned* __restrict__ p, int n, unsigned v) {
  int t = blockIdx.x * blockDim.x + threadIdx.x;
  if (t < n) p[t] = v;
}
__global__ void cvt_bf16(const float* __restrict__ in, unsigned short* __restrict__ out,
                         int n, int relu) {
  int t = blockIdx.x * blockDim.x + threadIdx.x;
  if (t >= n) return;
  float v = in[t];
  if (relu) v = v > 0.f ? v : 0.f;
  out[t] = f2bf(v);
}

// ---------- WMMA bf16 GEMM: C[M,N] = A[M,K] * B[K,N], fp32 accumulate ----------
// One wave computes a 16x64 C tile (4 accumulators, A fragment reused 4x).
// blockDim.x = 128 (4 waves); total waves = (M/16)*(N/64).
__global__ void __launch_bounds__(128)
wmma_gemm_bf16(const unsigned short* __restrict__ A,
               const unsigned short* __restrict__ B,
               float* __restrict__ C, int M, int N, int K) {
  const int lane   = threadIdx.x & 31;
  const int wid    = threadIdx.x >> 5;            // wave-uniform
  const int tilesN = N >> 6;                      // # of 64-wide wave tiles
  const int wave   = blockIdx.x * 4 + wid;
  const int m0     = (wave / tilesN) * 16;
  const int n0     = (wave % tilesN) * 64;
  if (m0 >= M) return;                            // wave-uniform: EXEC stays all-1s
  const int mr = lane & 15;
  const int hs = lane >> 4;
  const unsigned short* arow  = A + (size_t)(m0 + mr) * K;
  const unsigned short* bbase = B + n0;
  v8f acc0 = {}, acc1 = {}, acc2 = {}, acc3 = {};
  for (int k0 = 0; k0 < K; k0 += 32) {
    if (k0 + 32 < K) __builtin_prefetch(arow + k0 + 32, 0, 0); // global_prefetch_b8
    // A fragment: lane(m=mr, hs): elems 0..7 = K {8hs..8hs+7}, 8..15 = K {16+8hs..}
    v8bf alo = *reinterpret_cast<const v8bf*>(arow + k0 + 8 * hs);
    v8bf ahi = *reinterpret_cast<const v8bf*>(arow + k0 + 16 + 8 * hs);
    v16bf a;
#pragma unroll
    for (int i = 0; i < 8; ++i) { a[i] = alo[i]; a[i + 8] = ahi[i]; }
    // B fragments: lane(n=mr, hs): elems 0..7 = K {8hs..8hs+7}, 8..15 = K {16+8hs..}; 4 adjacent 16-col tiles
    const unsigned short* bcol = bbase + mr;
    v16bf b0, b1, b2, b3;
#pragma unroll
    for (int i = 0; i < 16; ++i) {
      const int kk = k0 + (i < 8 ? 8 * hs + i : 16 + 8 * hs + (i - 8));
      const __bf16* br = reinterpret_cast<const __bf16*>(bcol + (size_t)kk * N);
      b0[i] = br[0]; b1[i] = br[16]; b2[i] = br[32]; b3[i] = br[48];
    }
    acc0 = __builtin_amdgcn_wmma_f32_16x16x32_bf16(false, a, false, b0, (short)0, acc0, false, false);
    acc1 = __builtin_amdgcn_wmma_f32_16x16x32_bf16(false, a, false, b1, (short)0, acc1, false, false);
    acc2 = __builtin_amdgcn_wmma_f32_16x16x32_bf16(false, a, false, b2, (short)0, acc2, false, false);
    acc3 = __builtin_amdgcn_wmma_f32_16x16x32_bf16(false, a, false, b3, (short)0, acc3, false, false);
  }
  // D layout: element v -> C[m0 + v + 8*hs, n_tile + (lane&15)]
  float* crow = C + (size_t)(m0 + 8 * hs) * N + n0 + mr;
#pragma unroll
  for (int v = 0; v < 8; ++v) {
    float* r = crow + (size_t)v * N;
    r[0]  = acc0[v];
    r[16] = acc1[v];
    r[32] = acc2[v];
    r[48] = acc3[v];
  }
}

// ---------- degree / normalization ----------
__global__ void deg_scatter(const long long* __restrict__ ei, float* __restrict__ deg) {
  int e = blockIdx.x * blockDim.x + threadIdx.x;
  if (e < N_EDGES) atomicAdd(&deg[(int)ei[N_EDGES + e]], 1.0f);
}
__global__ void deg_rsqrt(float* __restrict__ deg) {   // in-place -> dinv
  int i = blockIdx.x * blockDim.x + threadIdx.x;
  if (i < N_NODES) deg[i] = rsqrtf(deg[i]);            // deg >= 1 (self loop)
}

// ---------- GCN aggregation ----------
template <int F>
__global__ void gcn_init(const float* __restrict__ h, const float* __restrict__ b,
                         const float* __restrict__ dinv, float* __restrict__ agg) {
  int t = blockIdx.x * blockDim.x + threadIdx.x;
  if (t >= N_NODES * F) return;
  int n = t / F, f = t % F;
  float di = dinv[n];
  agg[t] = b[f] + h[t] * di * di;                 // bias + self-loop message
}
template <int F>
__global__ void gcn_scatter(const long long* __restrict__ ei,
                            const float* __restrict__ dinv,
                            const float* __restrict__ h, float* __restrict__ agg) {
  constexpr int CH = F / 4;
  int t = blockIdx.x * blockDim.x + threadIdx.x;
  if (t >= N_EDGES * CH) return;
  int e = t / CH, c = t % CH;
  int s = (int)ei[e], d = (int)ei[N_EDGES + e];
  float coef = dinv[s] * dinv[d];
  const float* hs_ = h   + (size_t)s * F + c * 4;
  float*       od  = agg + (size_t)d * F + c * 4;
#pragma unroll
  for (int j = 0; j < 4; ++j) atomicAdd(&od[j], coef * hs_[j]);
}

// ---------- GAT attention ----------
__global__ void gat_al(const float* __restrict__ hg, const float* __restrict__ a_src,
                       const float* __restrict__ a_dst, float* __restrict__ als,
                       float* __restrict__ ald) {
  int t = blockIdx.x * blockDim.x + threadIdx.x;
  if (t >= N_NODES * HGAT) return;
  int n = t / HGAT, h = t % HGAT;
  const float* hv = hg + (size_t)n * HF + h * F_GAT;
  const float* as = a_src + h * F_GAT;
  const float* ad = a_dst + h * F_GAT;
  float s1 = 0.f, s2 = 0.f;
  for (int f = 0; f < F_GAT; ++f) { float v = hv[f]; s1 += v * as[f]; s2 += v * ad[f]; }
  als[t] = s1; ald[t] = s2;
}
__global__ void gat_att_max(const long long* __restrict__ ei,
                            const float* __restrict__ als, const float* __restrict__ ald,
                            float* __restrict__ ebuf, unsigned* __restrict__ mord) {
  int t = blockIdx.x * blockDim.x + threadIdx.x;
  if (t >= E_TOT * HGAT) return;
  int e = t / HGAT, h = t % HGAT, s, d;
  edge_sd(ei, e, s, d);
  float el = als[s * HGAT + h] + ald[d * HGAT + h];
  el = el > 0.f ? el : NEG_SLOPE * el;            // LeakyReLU(0.2)
  ebuf[t] = el;
  atomicMax(&mord[d * HGAT + h], f2ord(el));      // global_atomic_max_u32
}
__global__ void gat_att_exp(const long long* __restrict__ ei, float* __restrict__ ebuf,
                            const unsigned* __restrict__ mord, float* __restrict__ ssum) {
  int t = blockIdx.x * blockDim.x + threadIdx.x;
  if (t >= E_TOT * HGAT) return;
  int e = t / HGAT, h = t % HGAT, s, d;
  edge_sd(ei, e, s, d);
  float ex = __expf(ebuf[t] - ord2f(mord[d * HGAT + h]));
  ebuf[t] = ex;
  atomicAdd(&ssum[d * HGAT + h], ex);
}
__global__ void gat_att_agg(const long long* __restrict__ ei,
                            const float* __restrict__ ebuf, const float* __restrict__ ssum,
                            const float* __restrict__ hg, float* __restrict__ agg) {
  int t = blockIdx.x * blockDim.x + threadIdx.x;     // (edge, head, 4-float chunk)
  if (t >= E_TOT * HGAT * (F_GAT / 4)) return;
  int c = t & 31, h = (t >> 5) & 7, e = t >> 8, s, d;
  edge_sd(ei, e, s, d);
  float alpha = ebuf[e * HGAT + h] / (ssum[d * HGAT + h] + EPS_GAT);
  const float* hs_ = hg  + (size_t)s * HF + h * F_GAT + c * 4;
  float*       od  = agg + (size_t)d * HF + h * F_GAT + c * 4;
#pragma unroll
  for (int j = 0; j < 4; ++j) atomicAdd(&od[j], alpha * hs_[j]);
}

// ---------- pool + FC + softmax ----------
__global__ void pool_kernel(const float* __restrict__ agg, const float* __restrict__ bg,
                            float* __restrict__ pooled) {
  int f = blockIdx.x * blockDim.x + threadIdx.x;
  if (f >= HF) return;
  float b = bg[f], s = 0.f;
  for (int n = 0; n < N_NODES; ++n) {
    float v = agg[(size_t)n * HF + f] + b;
    s += v > 0.f ? v : 0.f;                       // ReLU before mean
  }
  pooled[f] = s * (1.0f / (float)N_NODES);
}
__global__ void fc_softmax(const float* __restrict__ pooled, const float* __restrict__ Wfc,
                           const float* __restrict__ bfc, float* __restrict__ out) {
  __shared__ float sm[NCLS];
  int c = threadIdx.x;
  if (c < NCLS) {
    float v = bfc[c];
    for (int k = 0; k < HF; ++k) v += pooled[k] * Wfc[k * NCLS + c];
    sm[c] = v;
  }
  __syncthreads();
  if (c == 0) {
    float mx = sm[0];
    for (int i = 1; i < NCLS; ++i) mx = fmaxf(mx, sm[i]);
    float e[NCLS], s = 0.f;
    for (int i = 0; i < NCLS; ++i) { e[i] = __expf(sm[i] - mx); s += e[i]; }
    float inv = 1.0f / s;
    for (int i = 0; i < NCLS; ++i) out[i] = e[i] * inv;
  }
}

// ---------- host launcher ----------
static inline dim3 gr(long long n, int b) { return dim3((unsigned)((n + b - 1) / b)); }

extern "C" void kernel_launch(void* const* d_in, const int* in_sizes, int n_in,
                              void* d_out, int out_size, void* d_ws, size_t ws_size,
                              hipStream_t stream) {
  const float*     x     = (const float*)d_in[0];
  const long long* ei    = (const long long*)d_in[1];   // int64 edge_index [2,E]
  const float*     W1    = (const float*)d_in[2];
  const float*     b1    = (const float*)d_in[3];
  const float*     W2    = (const float*)d_in[4];
  const float*     b2    = (const float*)d_in[5];
  const float*     Wg    = (const float*)d_in[6];
  const float*     a_src = (const float*)d_in[7];
  const float*     a_dst = (const float*)d_in[8];
  const float*     bg    = (const float*)d_in[9];
  const float*     Wfc   = (const float*)d_in[10];
  const float*     bfc   = (const float*)d_in[11];
  float* out = (float*)d_out;
  (void)in_sizes; (void)n_in; (void)out_size; (void)ws_size;

  // bump allocator over workspace (256B aligned)
  char* p = (char*)d_ws;
  auto alloc = [&](size_t bytes) -> char* {
    char* r = p; p += (bytes + 255) & ~(size_t)255; return r;
  };
  float*          dinv  = (float*)alloc((size_t)N_NODES * 4);
  float*          hbuf  = (float*)alloc((size_t)N_NODES * F_IN * 4);     // GEMM out
  float*          aggb  = (float*)alloc((size_t)N_NODES * F_IN * 4);     // GCN agg
  unsigned short* bfA   = (unsigned short*)alloc((size_t)N_NODES * F_IN * 2);
  unsigned short* bfW   = (unsigned short*)alloc((size_t)F_IN * HF * 2);
  float*          hg    = (float*)alloc((size_t)N_NODES * HF * 4);       // 80 MB
  float*          gagg  = (float*)alloc((size_t)N_NODES * HF * 4);       // 80 MB
  float*          als   = (float*)alloc((size_t)N_NODES * HGAT * 4);
  float*          ald   = (float*)alloc((size_t)N_NODES * HGAT * 4);
  unsigned*       mord  = (unsigned*)alloc((size_t)N_NODES * HGAT * 4);
  float*          ssum  = (float*)alloc((size_t)N_NODES * HGAT * 4);
  float*          ebuf  = (float*)alloc((size_t)E_TOT * HGAT * 4);
  float*          pooled= (float*)alloc((size_t)HF * 4);

  const int B = 256;

  // degrees (incl. self loop) -> dinv
  fill_f32<<<gr(N_NODES, B), B, 0, stream>>>(dinv, N_NODES, 1.0f);
  deg_scatter<<<gr(N_EDGES, B), B, 0, stream>>>(ei, dinv);
  deg_rsqrt<<<gr(N_NODES, B), B, 0, stream>>>(dinv);

  // ---- GCN layer 1: h1 = X*W1 ; agg ; (ReLU fused into next cvt) ----
  cvt_bf16<<<gr((size_t)N_NODES * F_IN, B), B, 0, stream>>>(x, bfA, N_NODES * F_IN, 0);
  cvt_bf16<<<gr(F_IN * 64, B), B, 0, stream>>>(W1, bfW, F_IN * 64, 0);
  {
    long long waves = (long long)(N_NODES / 16) * (64 / 64);
    wmma_gemm_bf16<<<gr(waves, 4), 128, 0, stream>>>(bfA, bfW, hbuf, N_NODES, 64, F_IN);
  }
  gcn_init<64><<<gr((size_t)N_NODES * 64, B), B, 0, stream>>>(hbuf, b1, dinv, aggb);
  gcn_scatter<64><<<gr((size_t)N_EDGES * 16, B), B, 0, stream>>>(ei, dinv, hbuf, aggb);

  // ---- GCN layer 2: h2 = relu(agg1)*W2 ; agg ----
  cvt_bf16<<<gr((size_t)N_NODES * 64, B), B, 0, stream>>>(aggb, bfA, N_NODES * 64, 1);
  cvt_bf16<<<gr(64 * 128, B), B, 0, stream>>>(W2, bfW, 64 * 128, 0);
  {
    long long waves = (long long)(N_NODES / 16) * (128 / 64);
    wmma_gemm_bf16<<<gr(waves, 4), 128, 0, stream>>>(bfA, bfW, hbuf, N_NODES, 128, 64);
  }
  gcn_init<128><<<gr((size_t)N_NODES * 128, B), B, 0, stream>>>(hbuf, b2, dinv, aggb);
  gcn_scatter<128><<<gr((size_t)N_EDGES * 32, B), B, 0, stream>>>(ei, dinv, hbuf, aggb);

  // ---- GAT: hg = relu(agg2)*Wg  (20000x128 @ 128x1024) ----
  cvt_bf16<<<gr((size_t)N_NODES * 128, B), B, 0, stream>>>(aggb, bfA, N_NODES * 128, 1);
  cvt_bf16<<<gr(F_IN * HF, B), B, 0, stream>>>(Wg, bfW, F_IN * HF, 0);
  {
    long long waves = (long long)(N_NODES / 16) * (HF / 64);
    wmma_gemm_bf16<<<gr(waves, 4), 128, 0, stream>>>(bfA, bfW, hg, N_NODES, HF, F_IN);
  }

  // attention logits per node/head
  gat_al<<<gr((size_t)N_NODES * HGAT, B), B, 0, stream>>>(hg, a_src, a_dst, als, ald);

  // segment softmax over incoming edges
  fill_u32<<<gr((size_t)N_NODES * HGAT, B), B, 0, stream>>>(mord, N_NODES * HGAT, 0u);
  fill_f32<<<gr((size_t)N_NODES * HGAT, B), B, 0, stream>>>(ssum, N_NODES * HGAT, 0.f);
  fill_f32<<<gr((size_t)N_NODES * HF, B), B, 0, stream>>>(gagg, N_NODES * HF, 0.f);
  gat_att_max<<<gr((size_t)E_TOT * HGAT, B), B, 0, stream>>>(ei, als, ald, ebuf, mord);
  gat_att_exp<<<gr((size_t)E_TOT * HGAT, B), B, 0, stream>>>(ei, ebuf, mord, ssum);
  gat_att_agg<<<gr((size_t)E_TOT * HGAT * (F_GAT / 4), B), B, 0, stream>>>(ei, ebuf, ssum,
                                                                           hg, gagg);

  // pool (+bg, ReLU) -> FC -> softmax
  pool_kernel<<<gr(HF, B), B, 0, stream>>>(gagg, bg, pooled);
  fc_softmax<<<1, 32, 0, stream>>>(pooled, Wfc, bfc, out);
}
